// CausalSelfAttention_82918638616672
// MI455X (gfx1250) — compile-verified, matrix operand order checked by reading, not run on hardware
//
#include <hip/hip_runtime.h>
#include <hip/hip_bf16.h>

#define DEV __device__ __forceinline__

typedef __attribute__((ext_vector_type(16))) __bf16       v16bf;
typedef __attribute__((ext_vector_type(8)))  float        v8f;
typedef __attribute__((ext_vector_type(8)))  unsigned int v8u;
typedef int v4i_vs __attribute__((__vector_size__(16)));   // matches builtin param

constexpr int BZ = 2, SQ = 2048, DM = 1024, NHD = 16, HDM = 64;
constexpr float ATT_SCALE = 0.125f; // 1/sqrt(64)

// ---------- gfx1250 async global->LDS copy (ASYNCcnt) with fallback ----------

#if defined(__gfx1250__) &&                                                    \
    __has_builtin(__builtin_amdgcn_global_load_async_to_lds_b128) &&           \
    __has_builtin(__builtin_amdgcn_s_wait_asynccnt)
#define HAVE_ASYNC_LDS 1
#else
#define HAVE_ASYNC_LDS 0
#endif

DEV void cp_b128_to_lds(__bf16* lds_dst, const __bf16* gsrc) {
#if HAVE_ASYNC_LDS
  __builtin_amdgcn_global_load_async_to_lds_b128(
      (__attribute__((address_space(1))) v4i_vs*)gsrc,
      (__attribute__((address_space(3))) v4i_vs*)lds_dst, 0, 0);
#else
  *(uint4*)lds_dst = *(const uint4*)gsrc;
#endif
}

DEV void cp_wait_all() {
#if HAVE_ASYNC_LDS
  __builtin_amdgcn_s_wait_asynccnt(0);
#endif
}

// ---------- helpers ----------

DEV __bf16 f2bf(float f) {
  unsigned int u = __builtin_bit_cast(unsigned int, f);
  u += 0x7FFFu + ((u >> 16) & 1u); // round-to-nearest-even
  unsigned short s = (unsigned short)(u >> 16);
  return __builtin_bit_cast(__bf16, s);
}

// 16-bit A/B fragment where the K dimension is contiguous in memory.
// Lane L holds line (lane&15), lane-group g = L>>4.
// Per ISA 7.12.2 (16-bit A 16x32): VGPR v<4 holds K = 2v+{0,1}+8g,
// v>=4 holds K = 16+2(v-4)+{0,1}+8g  ->  dword index (v<4 ? v : v+4)+4g.
DEV v16bf load_frag_k(const __bf16* mat, int ld, int line0, int k0) {
  int lane = threadIdx.x & 31;
  int g = lane >> 4;
  const unsigned int* row =
      (const unsigned int*)(mat + (size_t)(line0 + (lane & 15)) * ld + k0);
  v8u u;
#pragma unroll
  for (int v = 0; v < 8; ++v)
    u[v] = row[((v < 4) ? v : v + 4) + g * 4];
  return __builtin_bit_cast(v16bf, u);
}

// B fragment from a row-major K x N matrix (K strided by ld): lane walks
// down column n0 + (lane&15) with the same K pattern as above.
DEV v16bf load_frag_n(const __bf16* mat, int ld, int k0, int n0) {
  int lane = threadIdx.x & 31;
  int g = lane >> 4;
  const __bf16* col = mat + (size_t)k0 * ld + n0 + (lane & 15);
  v16bf f;
#pragma unroll
  for (int e = 0; e < 16; ++e) {
    int v = e >> 1, j = e & 1;
    int k = ((v < 4) ? 2 * v + j : 16 + 2 * (v - 4) + j) + g * 8;
    f[e] = col[(size_t)k * ld];
  }
  return f;
}

DEV v8f wmma_bf16(v16bf a, v16bf b, v8f c) {
  return __builtin_amdgcn_wmma_f32_16x16x32_bf16(false, a, false, b,
                                                 (short)0, c, false, false);
}

// ---------- fp32 -> bf16 conversion ----------

__global__ void cvt_f32_bf16(const float* __restrict__ in,
                             __bf16* __restrict__ out, long n) {
  long i = (long)blockIdx.x * blockDim.x + threadIdx.x;
  if (i < n) out[i] = f2bf(in[i]);
}

// ---------- bf16 WMMA GEMM:  C[MxN] = A[MxK] * B[KxN] + bias ----------
// 4 waves / block; block owns a 128x64 tile (wave w: rows w*32..w*32+31).
// The 32x64 B-slab per K-step is staged in LDS (async) and reused 4x.

template <bool OUT_BF16>
__global__ __launch_bounds__(128)
void gemm_bf16_wmma(const __bf16* __restrict__ A, const __bf16* __restrict__ Bm,
                    const float* __restrict__ bias, float* __restrict__ Cf,
                    __bf16* __restrict__ Cb, int M, int N, int K) {
  constexpr int LDB = 72; // 64 + 8 pad (rows stay 16B-aligned, spreads banks)
  __shared__ __bf16 ldsB[32 * LDB];

  int tid = threadIdx.x;
  int wave = tid >> 5, lane = tid & 31;
  int n0 = blockIdx.x * 64;
  int m0 = blockIdx.y * 128 + wave * 32;

  v8f acc[2][4];
#pragma unroll
  for (int r = 0; r < 2; ++r)
#pragma unroll
    for (int c = 0; c < 4; ++c) acc[r][c] = (v8f)0.0f;

  for (int k0 = 0; k0 < K; k0 += 32) {
    __syncthreads(); // previous K-step's LDS reads complete
    // stage B[k0..k0+31][n0..n0+63]: 256 16B-chunks, 2 per thread
#pragma unroll
    for (int i = 0; i < 2; ++i) {
      int chunk = tid + i * 128;
      int r = chunk >> 3, c = (chunk & 7) * 8;
      cp_b128_to_lds(&ldsB[r * LDB + c], Bm + (size_t)(k0 + r) * N + n0 + c);
    }
    if (k0 + 32 < K) // prefetch next A slab -> global_prefetch_b8
      __builtin_prefetch(A + (size_t)m0 * K + k0 + 32, 0, 1);

    v16bf a0 = load_frag_k(A, K, m0, k0);
    v16bf a1 = load_frag_k(A, K, m0 + 16, k0);

    cp_wait_all();
    __syncthreads(); // B slab visible to all 4 waves

#pragma unroll
    for (int c = 0; c < 4; ++c) {
      v16bf b = load_frag_n(ldsB, LDB, 0, c * 16);
      acc[0][c] = wmma_bf16(a0, b, acc[0][c]);
      acc[1][c] = wmma_bf16(a1, b, acc[1][c]);
    }
  }

  int g = lane >> 4, nl = lane & 15;
#pragma unroll
  for (int r = 0; r < 2; ++r)
#pragma unroll
    for (int c = 0; c < 4; ++c) {
      int n = n0 + c * 16 + nl;
      float bv = bias ? bias[n] : 0.0f;
#pragma unroll
      for (int v = 0; v < 8; ++v) {
        int m = m0 + r * 16 + v + 8 * g;
        float val = acc[r][c][v] + bv;
        if constexpr (OUT_BF16)
          Cb[(size_t)m * N + n] = f2bf(val);
        else
          Cf[(size_t)m * N + n] = val;
      }
    }
}

// ---------- flash attention forward ----------
// qkv: (B, S, 3D) bf16 row-major; Q/K/V are column offsets 0 / D / 2D.
// One wave per (b, h, 16-row q tile). K-tiles of 32 keys. The V tile is
// async-copied into LDS at the top of each iteration so the DMA overlaps
// the Q.K^T WMMAs and the softmax update. Running softmax stats live in
// C-layout (row m = v + 8g); row reductions are shfl_xor over 16 lanes.

__global__ __launch_bounds__(32)
void attn_fwd_wmma(const __bf16* __restrict__ qkv, __bf16* __restrict__ attb) {
  constexpr int LDV = 72;
  __shared__ __bf16 lds_v[32 * LDV];
  __shared__ __bf16 lds_p[16 * 32];

  int lane = threadIdx.x & 31;
  int g = lane >> 4, nl = lane & 15;
  int q0 = blockIdx.x * 16;
  int h = blockIdx.y;
  int b = blockIdx.z;

  const int ld = 3 * DM;
  const __bf16* qbase = qkv + (size_t)b * SQ * ld + h * HDM;
  const __bf16* kbase = qbase + DM;
  const __bf16* vbase = qbase + 2 * DM;

  v16bf aQ0 = load_frag_k(qbase, ld, q0, 0);
  v16bf aQ1 = load_frag_k(qbase, ld, q0, 32);

  v8f o[4];
#pragma unroll
  for (int c = 0; c < 4; ++c) o[c] = (v8f)0.0f;
  float m_run[8], l_run[8];
#pragma unroll
  for (int v = 0; v < 8; ++v) { m_run[v] = -1e30f; l_run[v] = 0.0f; }

  int nkt = (q0 + 16 + 31) / 32;
  for (int kt = 0; kt < nkt; ++kt) {
    int kk0 = kt * 32;

    // ---- kick off async V-tile copy (32 keys x 64 feats = 256 chunks) ----
#pragma unroll
    for (int i = 0; i < 8; ++i) {
      int chunk = i * 32 + lane;
      int r = chunk >> 3, c = (chunk & 7) * 8;
      cp_b128_to_lds(&lds_v[r * LDV + c],
                     vbase + (size_t)(kk0 + r) * ld + c);
    }

    // ---- S = Q K^T (16x32 scores, two 16x16 C-tiles) ----
    v8f s[2];
#pragma unroll
    for (int nc = 0; nc < 2; ++nc) {
      s[nc] = (v8f)0.0f;
      v16bf bk0 = load_frag_k(kbase, ld, kk0 + nc * 16, 0);
      v16bf bk1 = load_frag_k(kbase, ld, kk0 + nc * 16, 32);
      s[nc] = wmma_bf16(aQ0, bk0, s[nc]);
      s[nc] = wmma_bf16(aQ1, bk1, s[nc]);
    }

    // ---- scale + causal mask + online softmax ----
    float p0[8], p1[8], tmax[8];
#pragma unroll
    for (int v = 0; v < 8; ++v) {
      int qi = q0 + v + 8 * g;
      float x0 = s[0][v] * ATT_SCALE;
      float x1 = s[1][v] * ATT_SCALE;
      if (kk0 + nl > qi) x0 = -1e30f;
      if (kk0 + 16 + nl > qi) x1 = -1e30f;
      p0[v] = x0; p1[v] = x1;
      tmax[v] = fmaxf(x0, x1);
    }
#pragma unroll
    for (int off = 1; off < 16; off <<= 1)
#pragma unroll
      for (int v = 0; v < 8; ++v)
        tmax[v] = fmaxf(tmax[v], __shfl_xor(tmax[v], off, 32));

    float rescale[8];
#pragma unroll
    for (int v = 0; v < 8; ++v) {
      float mn = fmaxf(m_run[v], tmax[v]);
      rescale[v] = __expf(m_run[v] - mn);
      m_run[v] = mn;
    }
    float tsum[8];
#pragma unroll
    for (int v = 0; v < 8; ++v) {
      p0[v] = __expf(p0[v] - m_run[v]);
      p1[v] = __expf(p1[v] - m_run[v]);
      tsum[v] = p0[v] + p1[v];
    }
#pragma unroll
    for (int off = 1; off < 16; off <<= 1)
#pragma unroll
      for (int v = 0; v < 8; ++v)
        tsum[v] += __shfl_xor(tsum[v], off, 32);
#pragma unroll
    for (int v = 0; v < 8; ++v)
      l_run[v] = l_run[v] * rescale[v] + tsum[v];
#pragma unroll
    for (int c = 0; c < 4; ++c)
#pragma unroll
      for (int v = 0; v < 8; ++v) o[c][v] *= rescale[v];

    // ---- re-shape P (C-layout -> A-fragment layout) through LDS ----
    __syncthreads();
#pragma unroll
    for (int v = 0; v < 8; ++v) {
      lds_p[(v + 8 * g) * 32 + nl] = f2bf(p0[v]);
      lds_p[(v + 8 * g) * 32 + 16 + nl] = f2bf(p1[v]);
    }
    __syncthreads();
    v16bf aP = load_frag_k(lds_p, 32, 0, 0);

    // ---- O += P V (V fragments from the async-staged LDS tile) ----
    cp_wait_all();
#pragma unroll
    for (int c = 0; c < 4; ++c) {
      v16bf bv = load_frag_n(lds_v, LDV, 0, c * 16);
      o[c] = wmma_bf16(aP, bv, o[c]);
    }
  }

  // ---- normalize + write (B,S,D) bf16 ----
  float inv[8];
#pragma unroll
  for (int v = 0; v < 8; ++v) inv[v] = 1.0f / l_run[v];
#pragma unroll
  for (int c = 0; c < 4; ++c)
#pragma unroll
    for (int v = 0; v < 8; ++v) {
      int m = q0 + v + 8 * g;
      attb[(size_t)(b * SQ + m) * DM + h * HDM + c * 16 + nl] =
          f2bf(o[c][v] * inv[v]);
    }
}

// ---------- launcher ----------

extern "C" void kernel_launch(void* const* d_in, const int* in_sizes, int n_in,
                              void* d_out, int out_size, void* d_ws,
                              size_t ws_size, hipStream_t stream) {
  const float* x      = (const float*)d_in[0]; // (B,S,D)
  const float* w_qkv  = (const float*)d_in[1]; // (D,3D)
  const float* b_qkv  = (const float*)d_in[2]; // (3D)
  const float* w_proj = (const float*)d_in[3]; // (D,D)
  const float* b_proj = (const float*)d_in[4]; // (D)
  float* out = (float*)d_out;

  const size_t M = (size_t)BZ * SQ;            // 4096
  const size_t nX = M * DM;
  const size_t nWqkv = (size_t)DM * 3 * DM;
  const size_t nWproj = (size_t)DM * DM;
  const size_t nQKV = M * 3 * DM;

  __bf16* xb     = (__bf16*)d_ws;
  __bf16* wqkvb  = xb + nX;
  __bf16* wprojb = wqkvb + nWqkv;
  __bf16* qkvb   = wprojb + nWproj;
  __bf16* attb   = qkvb + nQKV;

  cvt_f32_bf16<<<dim3((unsigned)((nX + 255) / 256)), 256, 0, stream>>>(
      x, xb, (long)nX);
  cvt_f32_bf16<<<dim3((unsigned)((nWqkv + 255) / 256)), 256, 0, stream>>>(
      w_qkv, wqkvb, (long)nWqkv);
  cvt_f32_bf16<<<dim3((unsigned)((nWproj + 255) / 256)), 256, 0, stream>>>(
      w_proj, wprojb, (long)nWproj);

  // qkv = x @ w_qkv + b_qkv   (M=4096, N=3072, K=1024), bf16 out
  {
    dim3 grid(3 * DM / 64, (unsigned)(M / 128));
    gemm_bf16_wmma<true><<<grid, 128, 0, stream>>>(
        xb, wqkvb, b_qkv, nullptr, qkvb, (int)M, 3 * DM, DM);
  }

  // flash attention, bf16 out (B,S,D)
  {
    dim3 grid(SQ / 16, NHD, BZ);
    attn_fwd_wmma<<<grid, 32, 0, stream>>>(qkvb, attb);
  }

  // out = att @ w_proj + b_proj  (M=4096, N=1024, K=1024), f32 out
  {
    dim3 grid(DM / 64, (unsigned)(M / 128));
    gemm_bf16_wmma<false><<<grid, 128, 0, stream>>>(
        attb, wprojb, b_proj, out, nullptr, (int)M, DM, DM);
  }
}
